// MusicVAE_2800318677541
// MI455X (gfx1250) — matrix operand order repaired and checked by reading, not compiled, not run on hardware
//
#include <hip/hip_runtime.h>
#include <hip/hip_bf16.h>

// MusicVAE forward for MI455X (gfx1250, wave32).
// Persistent per-batch-tile LSTM blocks; gates via v_wmma_f32_16x16x32_bf16.
// Weights packed to bf16 once per launch (L2-resident, ~7.6MB << 192MB L2).
// v4: B-fragment addresses = SGPR base + one per-lane 32-bit byte offset +
//     constant IOFFSET (folded into global_load_b128) -> no per-fragment
//     address VALU. Laundered per-lane byte offset defeats LICM hoist->spill
//     at zero instruction cost. Ping-pong LDS; inline-exp tanh.

#define TSEQ  512
#define BATCH 256
#define IN    88
#define INP   96      // I padded to multiple of 32 for K-tiling
#define HID   512
#define ZDIM  256
#define G4H   2048    // 4*H

typedef __bf16 v16bf __attribute__((ext_vector_type(16)));
typedef __bf16 v8bf  __attribute__((ext_vector_type(8)));
typedef float  v8f   __attribute__((ext_vector_type(8)));

__device__ __forceinline__ float sigmoidf_(float x) { return 1.0f / (1.0f + __expf(-x)); }
// inline tanh via v_exp_f32 (avoids OCML tanhf register pressure / calls)
__device__ __forceinline__ float tanhf_(float x) {
  float e = __expf(2.0f * x);
  return (e - 1.0f) / (e + 1.0f);
}

// Opaque per-lane value: breaks loop-invariance (no LICM hoist -> no spill),
// emits zero instructions, keeps pointer provenance (global_load, not flat).
__device__ __forceinline__ int opaque_v(int v) {
  asm volatile("" : "+v"(v));
  return v;
}

// B-matrix fragment (K=32 x N=16): B[k][n] = W[j][k].
// Address = W (SGPR base) + boff (per-lane byte offset) + cofs (constant,
// folded into the instruction IOFFSET).  Layout: lanes 0-15 hold K=0..15 of
// column n=lane; lanes 16-31 hold K=16..31 of column n=lane-16.
__device__ __forceinline__ v16bf load_b_frag(const __bf16* __restrict__ W,
                                             int boff, int cofs) {
  const char* pb = (const char*)W + boff;
  v8bf lo = *(const v8bf*)(pb + cofs);
  v8bf hi = *(const v8bf*)(pb + cofs + 16);
  v16bf r;
#pragma unroll
  for (int i = 0; i < 8; ++i) { r[i] = lo[i]; r[i + 8] = hi[i]; }
  return r;
}

// A-matrix fragment (M=16 x K=32) from row-major LDS tile [16][Kp].
__device__ __forceinline__ v16bf load_a_frag(const __bf16* A, int Kp, int kbase, int lane) {
  int m   = lane & 15;
  int sel = (lane >> 4) & 1;
  const __bf16* p = A + m * Kp + kbase + sel * 16;
  v8bf lo = *(const v8bf*)p;
  v8bf hi = *(const v8bf*)(p + 8);
  v16bf r;
#pragma unroll
  for (int i = 0; i < 8; ++i) { r[i] = lo[i]; r[i + 8] = hi[i]; }
  return r;
}

__device__ __forceinline__ v8f wmma_bf16(v16bf a, v16bf b, v8f c) {
  return __builtin_amdgcn_wmma_f32_16x16x32_bf16(false, a, false, b, (short)0, c, false, false);
}

// ---------------------------------------------------------------------------
// Weight pack: f32 [Rin][Cin] -> bf16 [Rout][Cout], zero padded.
__global__ void pack_bf16_kernel(const float* __restrict__ in, __bf16* __restrict__ outp,
                                 int Rin, int Cin, int Rout, int Cout) {
  int idx = blockIdx.x * blockDim.x + threadIdx.x;
  int total = Rout * Cout;
  if (idx >= total) return;
  int r = idx / Cout, c = idx - r * Cout;
  float v = (r < Rin && c < Cin) ? in[(size_t)r * Cin + c] : 0.0f;
  outp[idx] = (__bf16)v;
}

// ---------------------------------------------------------------------------
// Encoder: blocks 0-15 forward dir, 16-31 backward dir; 16 batch rows each.
// 16 waves; wave w owns hidden columns [w*32, w*32+32) across all 4 gates.
// Ping-pong sh_h / sh_x: one barrier per timestep.
__global__ __launch_bounds__(512, 1)
void lstm_encoder_kernel(const float* __restrict__ x,
                         const __bf16* __restrict__ Wih_f, const __bf16* __restrict__ Whh_f,
                         const float* __restrict__ bias_f,
                         const __bf16* __restrict__ Wih_b, const __bf16* __restrict__ Whh_b,
                         const float* __restrict__ bias_b,
                         float* __restrict__ h_fwd, float* __restrict__ h_bwd) {
  __shared__ __bf16 sh_h[2][16 * HID];   // 32 KB
  __shared__ __bf16 sh_x[2][16 * INP];   // 6 KB
  __shared__ float  sh_bias[G4H];        // 8 KB

  const int tid  = threadIdx.x;
  const int lane = tid & 31;
  const int w    = tid >> 5;
  const int dir  = blockIdx.x >> 4;
  const int b0   = (blockIdx.x & 15) * 16;

  const __bf16* Wih = dir ? Wih_b : Wih_f;
  const __bf16* Whh = dir ? Whh_b : Whh_f;
  const float*  bias = dir ? bias_b : bias_f;
  float* hout = dir ? h_bwd : h_fwd;

  // precompute x-tile scatter offsets for this thread (<=3 elements)
  int xg[3], xl[3];
#pragma unroll
  for (int i = 0; i < 3; ++i) {
    int e = tid + i * 512;
    if (e < 16 * IN) {
      int r = e / IN, cc = e - r * IN;
      xg[i] = (b0 + r) * (TSEQ * IN) + cc;  // + tt*IN at use
      xl[i] = r * INP + cc;
    } else {
      xg[i] = -1; xl[i] = 0;
    }
  }

  for (int e = tid; e < G4H; e += 512) sh_bias[e] = bias[e];
  for (int e = tid; e < 16 * HID; e += 512) sh_h[0][e] = (__bf16)0.0f;
  for (int e = tid; e < 16 * INP; e += 512) { sh_x[0][e] = (__bf16)0.0f; sh_x[1][e] = (__bf16)0.0f; }
  __syncthreads();
  {  // prefill x tile for t = 0
    const int t0 = dir ? (TSEQ - 1) : 0;
#pragma unroll
    for (int i = 0; i < 3; ++i)
      if (xg[i] >= 0) sh_x[0][xl[i]] = (__bf16)x[(size_t)xg[i] + (size_t)t0 * IN];
  }
  __syncthreads();

  const int colb = w * 32;
  const int nl   = lane & 15;
  const int kk16 = (lane >> 4) << 4;               // 0 or 16 (elements)
  const int boff_ih0 = 2 * ((colb + nl) * INP + kk16);  // per-lane byte offset
  const int boff_hh0 = 2 * ((colb + nl) * HID + kk16);

  float breg[8];
#pragma unroll
  for (int q = 0; q < 8; ++q) {
    int gate = q >> 1, p = q & 1;
    breg[q] = sh_bias[gate * HID + colb + p * 16 + nl];
  }
  float creg[16];
#pragma unroll
  for (int i = 0; i < 16; ++i) creg[i] = 0.0f;
  const v8f zero8 = {0.f, 0.f, 0.f, 0.f, 0.f, 0.f, 0.f, 0.f};

  for (int t = 0; t < TSEQ; ++t) {
    const int cur = t & 1, nxt = cur ^ 1;
    const int boff_ih = opaque_v(boff_ih0);   // opaque per-iteration (free)
    const int boff_hh = opaque_v(boff_hh0);

    v8f acc[8];
#pragma unroll
    for (int q = 0; q < 8; ++q) acc[q] = zero8;

    // gates += x_t @ Wih^T   (K = 96, 3 K-tiles)
#pragma unroll
    for (int kt = 0; kt < 3; ++kt) {
      v16bf a = load_a_frag(sh_x[cur], INP, kt * 32, lane);
#pragma unroll
      for (int q = 0; q < 8; ++q) {
        int gate = q >> 1, p = q & 1;
        int cofs = 2 * ((gate * HID + p * 16) * INP + kt * 32);  // constant
        acc[q] = wmma_bf16(a, load_b_frag(Wih, boff_ih, cofs), acc[q]);
      }
    }
    // gates += h @ Whh^T   (K = 512, 16 K-tiles)
#pragma unroll 4
    for (int kt = 0; kt < 16; ++kt) {
      v16bf a = load_a_frag(sh_h[cur], HID, kt * 32, lane);
#pragma unroll
      for (int q = 0; q < 8; ++q) {
        int gate = q >> 1, p = q & 1;
        int cofs = 2 * ((gate * HID + p * 16) * HID + kt * 32);  // constant
        acc[q] = wmma_bf16(a, load_b_frag(Whh, boff_hh, cofs), acc[q]);
      }
    }

    // lane-local LSTM cell update -> write new h into the other buffer
#pragma unroll
    for (int p = 0; p < 2; ++p) {
#pragma unroll
      for (int r = 0; r < 8; ++r) {
        int m = (lane < 16) ? r : (r + 8);
        float iv = sigmoidf_(acc[0 + p][r] + breg[0 + p]);
        float fv = sigmoidf_(acc[2 + p][r] + breg[2 + p]);
        float gv = tanhf_  (acc[4 + p][r] + breg[4 + p]);
        float ov = sigmoidf_(acc[6 + p][r] + breg[6 + p]);
        float c = fv * creg[p * 8 + r] + iv * gv;
        creg[p * 8 + r] = c;
        float h = ov * tanhf_(c);
        sh_h[nxt][m * HID + colb + p * 16 + nl] = (__bf16)h;
      }
    }

    // prefetch next x tile into the other buffer
    if (t + 1 < TSEQ) {
      const int tt = dir ? (TSEQ - 2 - t) : (t + 1);
#pragma unroll
      for (int i = 0; i < 3; ++i)
        if (xg[i] >= 0) sh_x[nxt][xl[i]] = (__bf16)x[(size_t)xg[i] + (size_t)tt * IN];
    }
    __syncthreads();  // new h + next x visible; all reads of cur buffers done
  }

  const __bf16* hfin = sh_h[TSEQ & 1];
  for (int e = tid; e < 16 * HID; e += 512) {
    int r = e >> 9, c = e & 511;
    hout[(size_t)(b0 + r) * HID + c] = (float)hfin[r * HID + c];
  }
}

// ---------------------------------------------------------------------------
// Latent: mu/logvar/z (naive f32; ~0.3 GFLOP, negligible)
__global__ void latent_kernel(const float* __restrict__ hf, const float* __restrict__ hb,
                              const float* __restrict__ Wmu, const float* __restrict__ bmu,
                              const float* __restrict__ Wlv, const float* __restrict__ blv,
                              const float* __restrict__ eps,
                              float* __restrict__ mu_out, float* __restrict__ lv_out,
                              float* __restrict__ z_out) {
  int b = blockIdx.x, zc = threadIdx.x;
  const float* hfr = hf + (size_t)b * HID;
  const float* hbr = hb + (size_t)b * HID;
  const float* wm = Wmu + (size_t)zc * (2 * HID);
  const float* wl = Wlv + (size_t)zc * (2 * HID);
  float am = bmu[zc], al = blv[zc];
  for (int k = 0; k < HID; ++k) { float v = hfr[k]; am += v * wm[k];       al += v * wl[k]; }
  for (int k = 0; k < HID; ++k) { float v = hbr[k]; am += v * wm[HID + k]; al += v * wl[HID + k]; }
  size_t o = (size_t)b * ZDIM + zc;
  mu_out[o] = am;
  lv_out[o] = al;
  z_out[o]  = am + eps[o] * __expf(0.5f * al);
}

__global__ void h0_kernel(const float* __restrict__ z, const float* __restrict__ Wdi,
                          const float* __restrict__ bdi, float* __restrict__ h0) {
  int b = blockIdx.x, j = threadIdx.x;  // 512 threads
  const float* zr = z + (size_t)b * ZDIM;
  const float* wr = Wdi + (size_t)j * ZDIM;
  float a = bdi[j];
  for (int k = 0; k < ZDIM; ++k) a += zr[k] * wr[k];
  h0[(size_t)b * HID + j] = a;
}

// ---------------------------------------------------------------------------
// Decoder: 16 blocks x 16 batch rows; same gate scheme; output projection
// (waves 0-5) also via WMMA, feeding next-step input tile in LDS.
__global__ __launch_bounds__(512, 1)
void lstm_decoder_kernel(const float* __restrict__ h0,
                         const __bf16* __restrict__ Wih, const __bf16* __restrict__ Whh,
                         const float* __restrict__ bias,
                         const __bf16* __restrict__ Wo_p, const float* __restrict__ bo,
                         float* __restrict__ out) {
  __shared__ __bf16 sh_h[2][16 * HID];
  __shared__ __bf16 sh_in[2][16 * INP];
  __shared__ float  sh_bias[G4H];

  const int tid  = threadIdx.x;
  const int lane = tid & 31;
  const int w    = tid >> 5;
  const int b0   = blockIdx.x * 16;

  for (int e = tid; e < G4H; e += 512) sh_bias[e] = bias[e];
  for (int e = tid; e < 16 * INP; e += 512) { sh_in[0][e] = (__bf16)0.0f; sh_in[1][e] = (__bf16)0.0f; }
  for (int e = tid; e < 16 * HID; e += 512) {
    int r = e >> 9, c = e & 511;
    sh_h[0][e] = (__bf16)h0[(size_t)(b0 + r) * HID + c];
  }
  __syncthreads();

  const int colb = w * 32;
  const int nl   = lane & 15;
  const int kk16 = (lane >> 4) << 4;
  const int boff_ih0 = 2 * ((colb + nl) * INP + kk16);
  const int boff_hh0 = 2 * ((colb + nl) * HID + kk16);
  const int boff_wo0 = 2 * ((w * 16 + nl) * HID + kk16);  // waves 0-5 only

  float breg[8];
#pragma unroll
  for (int q = 0; q < 8; ++q) {
    int gate = q >> 1, p = q & 1;
    breg[q] = sh_bias[gate * HID + colb + p * 16 + nl];
  }
  float creg[16];
#pragma unroll
  for (int i = 0; i < 16; ++i) creg[i] = 0.0f;
  const int   oc    = w * 16 + nl;                       // output column (waves 0-5)
  const float boreg = (w < 6 && oc < IN) ? bo[oc] : 0.0f;
  const v8f zero8 = {0.f, 0.f, 0.f, 0.f, 0.f, 0.f, 0.f, 0.f};

  for (int t = 0; t < TSEQ; ++t) {
    const int cur = t & 1, nxt = cur ^ 1;
    const int boff_ih = opaque_v(boff_ih0);
    const int boff_hh = opaque_v(boff_hh0);

    v8f acc[8];
#pragma unroll
    for (int q = 0; q < 8; ++q) acc[q] = zero8;

#pragma unroll
    for (int kt = 0; kt < 3; ++kt) {
      v16bf a = load_a_frag(sh_in[cur], INP, kt * 32, lane);
#pragma unroll
      for (int q = 0; q < 8; ++q) {
        int gate = q >> 1, p = q & 1;
        int cofs = 2 * ((gate * HID + p * 16) * INP + kt * 32);
        acc[q] = wmma_bf16(a, load_b_frag(Wih, boff_ih, cofs), acc[q]);
      }
    }
#pragma unroll 4
    for (int kt = 0; kt < 16; ++kt) {
      v16bf a = load_a_frag(sh_h[cur], HID, kt * 32, lane);
#pragma unroll
      for (int q = 0; q < 8; ++q) {
        int gate = q >> 1, p = q & 1;
        int cofs = 2 * ((gate * HID + p * 16) * HID + kt * 32);
        acc[q] = wmma_bf16(a, load_b_frag(Whh, boff_hh, cofs), acc[q]);
      }
    }

    // lane-local LSTM update -> new h into other buffer (no WAR hazard)
#pragma unroll
    for (int p = 0; p < 2; ++p) {
#pragma unroll
      for (int r = 0; r < 8; ++r) {
        int m = (lane < 16) ? r : (r + 8);
        float iv = sigmoidf_(acc[0 + p][r] + breg[0 + p]);
        float fv = sigmoidf_(acc[2 + p][r] + breg[2 + p]);
        float gv = tanhf_  (acc[4 + p][r] + breg[4 + p]);
        float ov = sigmoidf_(acc[6 + p][r] + breg[6 + p]);
        float c = fv * creg[p * 8 + r] + iv * gv;
        creg[p * 8 + r] = c;
        float h = ov * tanhf_(c);
        sh_h[nxt][m * HID + colb + p * 16 + nl] = (__bf16)h;
      }
    }
    __syncthreads();  // new h visible (gates phase already consumed cur bufs)

    // pred = sigmoid(h_new @ Wo^T + bo): 6 N-tiles (96 cols, >=88 masked)
    if (w < 6) {
      const int boff_wo = opaque_v(boff_wo0);
      v8f pacc = zero8;
#pragma unroll 4
      for (int kt = 0; kt < 16; ++kt) {
        v16bf a = load_a_frag(sh_h[nxt], HID, kt * 32, lane);
        pacc = wmma_bf16(a, load_b_frag(Wo_p, boff_wo, 2 * (kt * 32)), pacc);
      }
#pragma unroll
      for (int r = 0; r < 8; ++r) {
        int m = (lane < 16) ? r : (r + 8);
        float v = sigmoidf_(pacc[r] + boreg);
        if (oc < IN) {
          out[((size_t)(b0 + m) * TSEQ + t) * IN + oc] = v;
          sh_in[nxt][m * INP + oc] = (__bf16)v;  // next-step input
        }
      }
    }
    __syncthreads();  // sh_in[nxt] ready for next step
  }
}

// ---------------------------------------------------------------------------
extern "C" void kernel_launch(void* const* d_in, const int* in_sizes, int n_in,
                              void* d_out, int out_size, void* d_ws, size_t ws_size,
                              hipStream_t stream) {
  (void)in_sizes; (void)n_in; (void)out_size; (void)ws_size;
  const float* x      = (const float*)d_in[0];
  const float* eps    = (const float*)d_in[1];
  const float* eWf_ih = (const float*)d_in[2];
  const float* eWf_hh = (const float*)d_in[3];
  const float* eb_f   = (const float*)d_in[4];
  const float* eWb_ih = (const float*)d_in[5];
  const float* eWb_hh = (const float*)d_in[6];
  const float* eb_b   = (const float*)d_in[7];
  const float* Wmu    = (const float*)d_in[8];
  const float* bmu    = (const float*)d_in[9];
  const float* Wlv    = (const float*)d_in[10];
  const float* blv    = (const float*)d_in[11];
  const float* Wdi    = (const float*)d_in[12];
  const float* bdi    = (const float*)d_in[13];
  const float* dW_ih  = (const float*)d_in[14];
  const float* dW_hh  = (const float*)d_in[15];
  const float* db     = (const float*)d_in[16];
  const float* Wo     = (const float*)d_in[17];
  const float* bo     = (const float*)d_in[18];

  float* out    = (float*)d_out;
  float* mu_out = out + (size_t)BATCH * TSEQ * IN;
  float* lv_out = mu_out + (size_t)BATCH * ZDIM;

  // workspace layout
  char* ws = (char*)d_ws;
  size_t off = 0;
  auto wsalloc = [&](size_t bytes) -> char* {
    char* p = ws + off;
    off = (off + bytes + 255) & ~(size_t)255;
    return p;
  };
  __bf16* Wf_ih_p = (__bf16*)wsalloc((size_t)G4H * INP * 2);
  __bf16* Wb_ih_p = (__bf16*)wsalloc((size_t)G4H * INP * 2);
  __bf16* dW_ih_p = (__bf16*)wsalloc((size_t)G4H * INP * 2);
  __bf16* Wf_hh_p = (__bf16*)wsalloc((size_t)G4H * HID * 2);
  __bf16* Wb_hh_p = (__bf16*)wsalloc((size_t)G4H * HID * 2);
  __bf16* dW_hh_p = (__bf16*)wsalloc((size_t)G4H * HID * 2);
  __bf16* Wo_p    = (__bf16*)wsalloc((size_t)INP * HID * 2);
  float*  h_fwd   = (float*)wsalloc((size_t)BATCH * HID * 4);
  float*  h_bwd   = (float*)wsalloc((size_t)BATCH * HID * 4);
  float*  z_buf   = (float*)wsalloc((size_t)BATCH * ZDIM * 4);
  float*  h0_buf  = (float*)wsalloc((size_t)BATCH * HID * 4);

  // pack weights f32 -> bf16 (padded)
  {
    int n = G4H * INP;
    pack_bf16_kernel<<<(n + 255) / 256, 256, 0, stream>>>(eWf_ih, Wf_ih_p, G4H, IN, G4H, INP);
    pack_bf16_kernel<<<(n + 255) / 256, 256, 0, stream>>>(eWb_ih, Wb_ih_p, G4H, IN, G4H, INP);
    pack_bf16_kernel<<<(n + 255) / 256, 256, 0, stream>>>(dW_ih,  dW_ih_p, G4H, IN, G4H, INP);
    int nh = G4H * HID;
    pack_bf16_kernel<<<(nh + 255) / 256, 256, 0, stream>>>(eWf_hh, Wf_hh_p, G4H, HID, G4H, HID);
    pack_bf16_kernel<<<(nh + 255) / 256, 256, 0, stream>>>(eWb_hh, Wb_hh_p, G4H, HID, G4H, HID);
    pack_bf16_kernel<<<(nh + 255) / 256, 256, 0, stream>>>(dW_hh,  dW_hh_p, G4H, HID, G4H, HID);
    int no = INP * HID;
    pack_bf16_kernel<<<(no + 255) / 256, 256, 0, stream>>>(Wo, Wo_p, IN, HID, INP, HID);
  }

  lstm_encoder_kernel<<<32, 512, 0, stream>>>(x, Wf_ih_p, Wf_hh_p, eb_f,
                                              Wb_ih_p, Wb_hh_p, eb_b, h_fwd, h_bwd);
  latent_kernel<<<BATCH, ZDIM, 0, stream>>>(h_fwd, h_bwd, Wmu, bmu, Wlv, blv, eps,
                                            mu_out, lv_out, z_buf);
  h0_kernel<<<BATCH, HID, 0, stream>>>(z_buf, Wdi, bdi, h0_buf);
  lstm_decoder_kernel<<<16, 512, 0, stream>>>(h0_buf, dW_ih_p, dW_hh_p, db, Wo_p, bo, out);
}
